// MultiHeadAttention_6485400617293
// MI455X (gfx1250) — matrix operand order repaired and checked by reading, not compiled, not run on hardware
//
#include <hip/hip_runtime.h>

// ---------------------------------------------------------------------------
// MultiHeadAttention forward for gfx1250 (MI455X), f16 WMMA + fp32 softmax.
// B=2, S=2048, H=1024, NH=16, HD=64. Inputs fp32, outputs fp32.
// d_in order (setup_inputs dict): x, Wq, bq, Wk, bk, Wv, bv, Wo, bo
// d_out: out [B*S*H] fp32, then avg_attention [B*S] fp32.
//
// Pipeline:
//   0) cvt_kernel x5 : fp32 -> f16 copies of x and the four weights (one-time,
//      bandwidth-trivial) so GEMM hot loops are pure b128 loads, no VALU cvt.
//   1) qkv_kernel    : fused Q/K/V projection, 12 WMMAs per k-step per wave.
//   2) attn_kernel   : per (b,h,16-query tile): QK^T via WMMA -> f16 LDS,
//      in-place wave softmax, PV via WMMA, avg-attention atomics.
//      LDS ~66KB -> 4 blocks/WGP occupancy.
//   3) oproj_kernel  : output projection + bias, fp32 out.
// ---------------------------------------------------------------------------

typedef _Float16 f16;
typedef __attribute__((ext_vector_type(16))) _Float16 v16h;
typedef __attribute__((ext_vector_type(8)))  float    v8f;
typedef unsigned short u16;
typedef unsigned int   u32;

constexpr int Bv = 2;
constexpr int Sv = 2048;
constexpr int Hv = 1024;
constexpr int NHv = 16;
constexpr int HDv = 64;
constexpr float SCALE = 0.125f; // HD^-0.5

// ---------------- helpers ----------------

__device__ inline u16 f2hraw(float f) {
    f16 h = (f16)f;
    return __builtin_bit_cast(u16, h);
}
__device__ inline float h2f(u16 h) {
    return (float)__builtin_bit_cast(f16, h);
}

__device__ inline v8f vzero8() {
    v8f v = {0.f, 0.f, 0.f, 0.f, 0.f, 0.f, 0.f, 0.f};
    return v;
}

struct U128x2 { uint4 a, b; };

// B-fragment (32x16 f16): 16 contiguous f16 per lane.
__device__ inline v16h loadB_f16(const u16* p) {
    U128x2 s;
    s.a = *(const uint4*)p;
    s.b = *(const uint4*)(p + 8);
    return __builtin_bit_cast(v16h, s);
}
// A-fragment (16x32 f16): 16 contiguous f16 per lane.
__device__ inline v16h loadA_f16(const u16* p) {
    U128x2 s;
    s.a = *(const uint4*)p;
    s.b = *(const uint4*)(p + 8);
    return __builtin_bit_cast(v16h, s);
}

__device__ inline v8f wmma_f16(v16h a, v16h b, v8f c) {
    return __builtin_amdgcn_wmma_f32_16x16x32_f16(false, a, false, b, (short)0, c,
                                                  false, false);
}

// ---------------------------------------------------------------------------
// Kernel 0: one-time fp32 -> f16 conversion (vectorized 8 elems/thread).
// ---------------------------------------------------------------------------
__global__ __launch_bounds__(256) void cvt_kernel(const float* __restrict__ in,
                                                  u16* __restrict__ out, int n8) {
    const int i = blockIdx.x * 256 + threadIdx.x;
    if (i >= n8) return;
    const float4* p = (const float4*)in + (size_t)i * 2;
    float4 a = p[0], b = p[1];
    uint4 o;
    o.x = (u32)f2hraw(a.x) | ((u32)f2hraw(a.y) << 16);
    o.y = (u32)f2hraw(a.z) | ((u32)f2hraw(a.w) << 16);
    o.z = (u32)f2hraw(b.x) | ((u32)f2hraw(b.y) << 16);
    o.w = (u32)f2hraw(b.z) | ((u32)f2hraw(b.w) << 16);
    ((uint4*)out)[i] = o;
}

// ---------------------------------------------------------------------------
// Kernel 1: fused Q/K/V projections (all-f16 operands).
// Each wave: rows [rt*16, rt*16+16), cols [cg*64, cg*64+64) of Q, K and V.
// y[m,n] = sum_h x[m,h] * W[n,h] + b[n].
// ---------------------------------------------------------------------------
__global__ __launch_bounds__(128) void qkv_kernel(
    const u16* __restrict__ xh,
    const u16* __restrict__ Wq, const u16* __restrict__ Wk,
    const u16* __restrict__ Wv,
    const float* __restrict__ bq, const float* __restrict__ bk,
    const float* __restrict__ bv,
    u16* __restrict__ Qh, u16* __restrict__ Kh, u16* __restrict__ Vt)
{
    const int lane = threadIdx.x & 31;
    const int l16 = lane & 15;
    const int half = lane >> 4;
    const int gwave = blockIdx.x * 4 + (threadIdx.x >> 5);
    const int rt = gwave >> 4;  // 0..255 (B*S/16 row tiles)
    const int cg = gwave & 15;  // 0..15  (H/64 col groups)
    const int m0 = rt * 16;
    const int n0 = cg * 64;

    const u16* Ws[3] = {Wq, Wk, Wv};
    const float* bs[3] = {bq, bk, bv};

    v8f acc[3][4];
#pragma unroll
    for (int t = 0; t < 3; ++t)
#pragma unroll
        for (int j = 0; j < 4; ++j) acc[t][j] = vzero8();

    const u16* xrow = xh + (size_t)(m0 + l16) * Hv;
    for (int kk = 0; kk < 32; ++kk) {
        const int k0 = kk * 32;
        v16h afrag = loadA_f16(xrow + k0 + half * 16);
#pragma unroll
        for (int t = 0; t < 3; ++t) {
            v16h bf[4];
#pragma unroll
            for (int j = 0; j < 4; ++j) {
                const int ncol = n0 + j * 16 + l16;
                bf[j] = loadB_f16(Ws[t] + (size_t)ncol * Hv + k0 + half * 16);
            }
#pragma unroll
            for (int j = 0; j < 4; ++j)
                acc[t][j] = wmma_f16(afrag, bf[j], acc[t][j]);
        }
    }

#pragma unroll
    for (int t = 0; t < 3; ++t) {
#pragma unroll
        for (int j = 0; j < 4; ++j) {
            const int ncol = n0 + j * 16 + l16;
            const float bias = bs[t][ncol];
            const int hh = ncol >> 6;   // head
            const int dd = ncol & 63;   // dim within head
#pragma unroll
            for (int r = 0; r < 8; ++r) {
                const int m = m0 + r + half * 8;
                const int bb = m >> 11;       // / S
                const int ss = m & (Sv - 1);  // % S
                const u16 hv = f2hraw(acc[t][j][r] + bias);
                const size_t bh = (size_t)bb * NHv + hh;
                if (t == 0)      Qh[(bh * Sv + ss) * HDv + dd] = hv;
                else if (t == 1) Kh[(bh * Sv + ss) * HDv + dd] = hv;
                else             Vt[(bh * HDv + dd) * Sv + ss] = hv;  // transposed
            }
        }
    }
}

// ---------------------------------------------------------------------------
// Kernel 2: attention. One block (4 waves) per (b, h, 16-query tile).
// LDS: scores/probs f16 [16][2048] (65536 B) + Q tile f16 [16][64] (2048 B)
//   = 67584 B -> 4 blocks per WGP (320 KB LDS).
// Softmax is done IN PLACE on the f16 buffer (same element size).
// ---------------------------------------------------------------------------
constexpr size_t LDS_SS = (size_t)16 * Sv * 2;
constexpr size_t LDS_SQ = (size_t)16 * HDv * 2;
constexpr size_t ATTN_LDS = LDS_SS + LDS_SQ;

__global__ __launch_bounds__(128) void attn_kernel(
    const u16* __restrict__ Qh, const u16* __restrict__ Kh,
    const u16* __restrict__ Vt, u16* __restrict__ ctx,
    float* __restrict__ avg)
{
    extern __shared__ char smem[];
    u16* sS = (u16*)smem;              // 16 x 2048 f16 (scores -> exp -> probs)
    u16* sQ = (u16*)(smem + LDS_SS);   // 16 x 64 f16

    const int tid = threadIdx.x;
    const int lane = tid & 31;
    const int wave = tid >> 5;
    const int l16 = lane & 15;
    const int half = lane >> 4;

    const int qt = blockIdx.x & 127;
    const int hh = (blockIdx.x >> 7) & 15;
    const int bb = blockIdx.x >> 11;
    const size_t bh = (size_t)bb * NHv + hh;

    // --- load Q tile (16x64 f16 = 1024 elems = 128 x uint4) ---
    const u16* Qg = Qh + (bh * Sv + (size_t)qt * 16) * HDv;
    ((uint4*)sQ)[tid] = ((const uint4*)Qg)[tid];
    __syncthreads();

    v16h aq0 = loadA_f16(sQ + l16 * HDv + 0 + half * 16);
    v16h aq1 = loadA_f16(sQ + l16 * HDv + 32 + half * 16);

    // --- scores: S_tile[16][2048] = Q @ K^T * scale, stored f16 ---
    const u16* Kg = Kh + bh * Sv * HDv;
    for (int kt = wave; kt < Sv / 16; kt += 4) {
        const int key = kt * 16 + l16;
        const u16* kp = Kg + (size_t)key * HDv + half * 16;
        v16h b0 = loadB_f16(kp);       // d = 0..31 chunk
        v16h b1 = loadB_f16(kp + 32);  // d = 32..63 chunk
        v8f c = vzero8();
        c = wmma_f16(aq0, b0, c);
        c = wmma_f16(aq1, b1, c);
#pragma unroll
        for (int r = 0; r < 8; ++r)
            sS[(r + half * 8) * Sv + kt * 16 + l16] = f2hraw(c[r] * SCALE);
    }
    __syncthreads();

    // --- softmax (in place, one wave per 4 rows) ---
    for (int row = wave * 4; row < wave * 4 + 4; ++row) {
        u16* srow = sS + (size_t)row * Sv;
        float mx = -1e30f;
        for (int i = lane; i < Sv; i += 32) mx = fmaxf(mx, h2f(srow[i]));
#pragma unroll
        for (int off = 16; off >= 1; off >>= 1)
            mx = fmaxf(mx, __shfl_xor(mx, off, 32));
        float sum = 0.f;
        for (int i = lane; i < Sv; i += 32) {
            float e = __expf(h2f(srow[i]) - mx);  // e in (0,1], exact f16 range
            srow[i] = f2hraw(e);
            sum += e;
        }
#pragma unroll
        for (int off = 16; off >= 1; off >>= 1)
            sum += __shfl_xor(sum, off, 32);
        const float inv = 1.0f / sum;  // butterfly leaves total in all lanes
        for (int i = lane; i < Sv; i += 32)
            srow[i] = f2hraw(h2f(srow[i]) * inv);
    }
    __syncthreads();

    // --- avg_attention: avg[b,k] += sum_q p[q,k] / (NH*S) ---
    const float norm = 1.0f / (float)(NHv * Sv);
    for (int k = tid; k < Sv; k += 128) {
        float s = 0.f;
#pragma unroll
        for (int r = 0; r < 16; ++r) s += h2f(sS[(size_t)r * Sv + k]);
        atomicAdd(&avg[(size_t)bb * Sv + k], s * norm);
    }

    // --- ctx = P @ V : wave owns 16 d-columns; pure b128 loads both sides ---
    const int dcol = wave * 16 + l16;
    const u16* vrow = Vt + (bh * HDv + dcol) * Sv;
    const u16* prow = sS + (size_t)l16 * Sv;
    v8f c = vzero8();
    for (int c0 = 0; c0 < Sv; c0 += 32) {
        v16h a = loadA_f16(prow + c0 + half * 16);
        v16h b = loadB_f16(vrow + c0 + half * 16);
        c = wmma_f16(a, b, c);
    }
#pragma unroll
    for (int r = 0; r < 8; ++r) {
        const int m = r + half * 8;
        const size_t sidx = (size_t)qt * 16 + m;
        ctx[((size_t)bb * Sv + sidx) * Hv + hh * HDv + dcol] = f2hraw(c[r]);
    }
}

// ---------------------------------------------------------------------------
// Kernel 3: output projection. out[m,n] = sum_h ctx[m,h]*Wo[n,h] + bo[n].
// ---------------------------------------------------------------------------
__global__ __launch_bounds__(128) void oproj_kernel(
    const u16* __restrict__ ctx, const u16* __restrict__ Wo,
    const float* __restrict__ bo, float* __restrict__ out)
{
    const int lane = threadIdx.x & 31;
    const int l16 = lane & 15;
    const int half = lane >> 4;
    const int gwave = blockIdx.x * 4 + (threadIdx.x >> 5);
    const int rt = gwave >> 4;
    const int cg = gwave & 15;
    const int m0 = rt * 16;
    const int n0 = cg * 64;

    v8f acc[4];
#pragma unroll
    for (int j = 0; j < 4; ++j) acc[j] = vzero8();

    const u16* crow = ctx + (size_t)(m0 + l16) * Hv;
    for (int kk = 0; kk < 32; ++kk) {
        const int k0 = kk * 32;
        v16h afrag = loadA_f16(crow + k0 + half * 16);
        v16h bf[4];
#pragma unroll
        for (int j = 0; j < 4; ++j) {
            const int ncol = n0 + j * 16 + l16;
            bf[j] = loadB_f16(Wo + (size_t)ncol * Hv + k0 + half * 16);
        }
#pragma unroll
        for (int j = 0; j < 4; ++j)
            acc[j] = wmma_f16(afrag, bf[j], acc[j]);
    }

#pragma unroll
    for (int j = 0; j < 4; ++j) {
        const int ncol = n0 + j * 16 + l16;
        const float bias = bo[ncol];
#pragma unroll
        for (int r = 0; r < 8; ++r) {
            const int m = m0 + r + half * 8;
            out[(size_t)m * Hv + ncol] = acc[j][r] + bias;
        }
    }
}

// ---------------------------------------------------------------------------
extern "C" void kernel_launch(void* const* d_in, const int* in_sizes, int n_in,
                              void* d_out, int out_size, void* d_ws, size_t ws_size,
                              hipStream_t stream) {
    (void)in_sizes; (void)n_in; (void)out_size; (void)ws_size;
    const float* x  = (const float*)d_in[0];
    const float* Wq = (const float*)d_in[1];
    const float* bq = (const float*)d_in[2];
    const float* Wk = (const float*)d_in[3];
    const float* bk = (const float*)d_in[4];
    const float* Wv = (const float*)d_in[5];
    const float* bv = (const float*)d_in[6];
    const float* Wo = (const float*)d_in[7];
    const float* bo = (const float*)d_in[8];

    float* out = (float*)d_out;
    float* avg = out + (size_t)Bv * Sv * Hv;

    const size_t nqkv = (size_t)Bv * NHv * Sv * HDv;  // 4M elems
    const size_t nx   = (size_t)Bv * Sv * Hv;         // 4M elems
    const size_t nw   = (size_t)Hv * Hv;              // 1M elems
    u16* Qh   = (u16*)d_ws;
    u16* Kh   = Qh + nqkv;
    u16* Vt   = Kh + nqkv;
    u16* ctx  = Vt + nqkv;    // [B,S,H] f16
    u16* xh   = ctx + nx;     // [B*S,H] f16
    u16* Wqh  = xh + nx;
    u16* Wkh  = Wqh + nw;
    u16* Wvh  = Wkh + nw;
    u16* Woh  = Wvh + nw;

    // 0) one-time fp32 -> f16 copies (bandwidth-trivial)
    cvt_kernel<<<dim3((nx / 8 + 255) / 256), dim3(256), 0, stream>>>(x, xh, (int)(nx / 8));
    cvt_kernel<<<dim3((nw / 8 + 255) / 256), dim3(256), 0, stream>>>(Wq, Wqh, (int)(nw / 8));
    cvt_kernel<<<dim3((nw / 8 + 255) / 256), dim3(256), 0, stream>>>(Wk, Wkh, (int)(nw / 8));
    cvt_kernel<<<dim3((nw / 8 + 255) / 256), dim3(256), 0, stream>>>(Wv, Wvh, (int)(nw / 8));
    cvt_kernel<<<dim3((nw / 8 + 255) / 256), dim3(256), 0, stream>>>(Wo, Woh, (int)(nw / 8));

    // 1) fused QKV projections (4096 waves -> 1024 blocks of 4 waves)
    qkv_kernel<<<dim3(1024), dim3(128), 0, stream>>>(xh, Wqh, Wkh, Wvh,
                                                     bq, bk, bv, Qh, Kh, Vt);
    // 2) zero avg-attention accumulator (graph-capture safe)
    hipMemsetAsync(avg, 0, (size_t)Bv * Sv * sizeof(float), stream);
    // 3) attention: one block per (b, h, 16-query tile) = 4096 blocks
    attn_kernel<<<dim3(Bv * NHv * (Sv / 16)), dim3(128), ATTN_LDS, stream>>>(
        Qh, Kh, Vt, ctx, avg);
    // 4) output projection
    oproj_kernel<<<dim3(1024), dim3(128), 0, stream>>>(ctx, Woh, bo, out);
}
